// MIMOSS_55009941127824
// MI455X (gfx1250) — matrix operand order repaired and checked by reading, not compiled, not run on hardware
//
#include <hip/hip_runtime.h>

// ---------------- CDNA5 WMMA types / helpers (wave32) ----------------
typedef __attribute__((ext_vector_type(8)))  __bf16 v8bf;
typedef __attribute__((ext_vector_type(16))) __bf16 v16bf;
typedef __attribute__((ext_vector_type(8)))  float  v8f;
typedef __attribute__((ext_vector_type(4)))  int    v4i;

typedef __attribute__((address_space(1))) void gas_void;  // global
typedef __attribute__((address_space(3))) void las_void;  // LDS
typedef __attribute__((address_space(1))) v4i  gas_v4i;   // global int4
typedef __attribute__((address_space(3))) v4i  las_v4i;   // LDS int4

#define B_      64
#define SEQ_    16384
#define ND      128      // state order
#define CHL     32       // chunk length (time steps)
#define UC      192      // 6*CHL : u-chunk / y-chunk width
#define CPS     8        // chunks per segment
#define NSEG    64       // segments (SEQ / (CHL*CPS))
#define NCHUNK  512      // SEQ / CHL

__device__ __forceinline__ v8f wmma_bf16(v16bf a, v16bf b, v8f c) {
  // D = A(16x32 bf16) x B(32x16 bf16) + C(16x16 f32)
  return __builtin_amdgcn_wmma_f32_16x16x32_bf16(
      false, a, false, b, (short)0, c, false, false);
}

// ---- async global->LDS copy (ASYNCcnt-tracked), 16B granules ----
__device__ __forceinline__ void async_b128(__bf16* lds_dst, const __bf16* g_src) {
#if __has_builtin(__builtin_amdgcn_global_load_async_to_lds_b128)
  __builtin_amdgcn_global_load_async_to_lds_b128(
      (gas_v4i*)(gas_void*)g_src, (las_v4i*)(las_void*)lds_dst, 0, 0);
#else
  unsigned ldso = (unsigned)(unsigned long long)(las_void*)lds_dst;
  unsigned long long ga = (unsigned long long)g_src;
  asm volatile("global_load_async_to_lds_b128 %0, %1, off"
               :: "v"(ldso), "v"(ga) : "memory");
#endif
}

template <int N>
__device__ __forceinline__ void wait_async() {
#if __has_builtin(__builtin_amdgcn_s_wait_asynccnt)
  __builtin_amdgcn_s_wait_asynccnt(N);
#else
  asm volatile("s_wait_asynccnt %0" :: "i"(N) : "memory");
#endif
}

// copy one 64x192 bf16 chunk (24576 B) : 256 threads x 6 b128 each
__device__ __forceinline__ void async_copy_chunk(__bf16* dst_lds,
                                                 const __bf16* src_glob, int tid) {
#pragma unroll
  for (int k = 0; k < 6; ++k) {
    int off = (tid + 256 * k) * 8;   // bf16 elements, 16B granule
    async_b128(dst_lds + off, src_glob + off);
  }
}

// A-fragment: 16x32 tile from row-major [M][K]; lane lm = M row,
// elements 0..7 -> K = lh*8 + 0..7 ; 8..15 -> K = 16 + lh*8 + 0..7
__device__ __forceinline__ v16bf load_a_frag(const __bf16* tile, int ld, int lane) {
  int lm = lane & 15, lh = lane >> 4;
  const __bf16* p = tile + lm * ld + lh * 8;
  v8bf lo = *(const v8bf*)(p);
  v8bf hi = *(const v8bf*)(p + 16);
  v16bf a;
#pragma unroll
  for (int i = 0; i < 8; ++i) { a[i] = lo[i]; a[i + 8] = hi[i]; }
  return a;
}

// B-fragment (K x 16) for GEMM  D = X * Op^T, with operator Op stored
// row-major [N][K]: lane column n = ln, elements 0..7 -> K = lh*8 + 0..7 ; 8..15 -> K = 16 + lh*8 + 0..7.
__device__ __forceinline__ v16bf load_b_frag(const __bf16* tile, int ld, int lane) {
  int ln = lane & 15, lh = lane >> 4;
  const __bf16* p = tile + ln * ld + lh * 8;
  v8bf lo = *(const v8bf*)(p);
  v8bf hi = *(const v8bf*)(p + 16);
  v16bf b;
#pragma unroll
  for (int i = 0; i < 8; ++i) { b[i] = lo[i]; b[i + 8] = hi[i]; }
  return b;
}

// ---------------- Phase 0: standardize input into chunked bf16 layout ----------------
// Ubf[chunk][batch][UC], UC index = t'*6 + i
__global__ void prep_u_kernel(const float* __restrict__ in,
                              const float* __restrict__ imean,
                              const float* __restrict__ istd,
                              __bf16* __restrict__ Ubf) {
  const int total = B_ * SEQ_ * 6;
  for (int e = blockIdx.x * blockDim.x + threadIdx.x; e < total;
       e += gridDim.x * blockDim.x) {
    int i  = e % 6;
    int bt = e / 6;
    int t  = bt % SEQ_;
    int b  = bt / SEQ_;
    float v = (in[e] - imean[i]) / istd[i];
    int chunk = t >> 5, tp = t & 31;
    Ubf[(size_t)(chunk * B_ + b) * UC + tp * 6 + i] = (__bf16)v;
  }
}

// ---------------- Phase 0: fp32 128x128 matmul (matrix-power squarings) ----------------
__global__ void mm128_kernel(float* __restrict__ D,
                             const float* __restrict__ A,
                             const float* __restrict__ Bm) {
  int idx = blockIdx.x * 256 + threadIdx.x;   // 64 blocks x 256 = 16384
  int r = idx >> 7, c = idx & 127;
  float s = 0.f;
  for (int k = 0; k < ND; ++k) s += A[r * ND + k] * Bm[k * ND + c];
  D[idx] = s;
}

// ---------------- Phase 0: build Cbar/Mbar/Gbar (fp32), one workgroup ----------------
// Cbar[6k+o][n] = (C A^k)[o][n]          (UC x ND)
// Mbar[n][6j+i] = (A^(31-j) B)[n][i]     (ND x UC)
// Gbar[6k+o][6j+i] = C A^(k-1-j) B (j<k), D (j==k), 0 (j>k)   (UC x UC)
__global__ void build_ops_kernel(const float* __restrict__ A,
                                 const float* __restrict__ Bm,   // [128][6]
                                 const float* __restrict__ Cm,   // [6][128]
                                 const float* __restrict__ Dm,   // [6][6]
                                 float* __restrict__ Cbar,
                                 float* __restrict__ Mbar,
                                 float* __restrict__ Gbar) {
  __shared__ float R[6 * ND], Rn[6 * ND];    // C A^k
  __shared__ float Mc[ND * 6], Mcn[ND * 6];  // A^k B
  __shared__ float cab[36];
  int tid = threadIdx.x;

  for (int e = tid; e < 6 * ND; e += 256) { R[e] = Cm[e]; Mc[e] = Bm[e]; }
  for (int e = tid; e < UC * UC; e += 256) Gbar[e] = 0.f;
  __syncthreads();
  for (int e = tid; e < CHL * 36; e += 256) {
    int k = e / 36, oi = e % 36, o = oi / 6, i = oi % 6;
    Gbar[(6 * k + o) * UC + 6 * k + i] = Dm[oi];
  }
  __syncthreads();

  for (int k = 0; k < CHL; ++k) {
    for (int e = tid; e < 6 * ND; e += 256) Cbar[k * 6 * ND + e] = R[e];
    for (int e = tid; e < ND * 6; e += 256) {
      int n = e / 6, i = e % 6;
      Mbar[n * UC + 6 * (CHL - 1 - k) + i] = Mc[e];
    }
    if (tid < 36) {   // cab = (C A^k) B   (6x6)
      int o = tid / 6, i = tid % 6;
      float s = 0.f;
      for (int n = 0; n < ND; ++n) s += R[o * ND + n] * Bm[n * 6 + i];
      cab[tid] = s;
    }
    __syncthreads();
    for (int e = tid; e < (CHL - 1 - k) * 36; e += 256) {
      int bkk = e / 36, oi = e % 36, o = oi / 6, i = oi % 6;
      int kk = k + 1 + bkk;                      // row block
      Gbar[(6 * kk + o) * UC + 6 * (kk - 1 - k) + i] = cab[oi];
    }
    if (k < CHL - 1) {
      for (int e = tid; e < 6 * ND; e += 256) {  // Rn = R * A
        int o = e / ND, n = e % ND;
        float s = 0.f;
        for (int m = 0; m < ND; ++m) s += R[o * ND + m] * A[m * ND + n];
        Rn[e] = s;
      }
      for (int e = tid; e < ND * 6; e += 256) {  // Mcn = A * Mc
        int n = e / 6, i = e % 6;
        float s = 0.f;
        for (int m = 0; m < ND; ++m) s += A[n * ND + m] * Mc[m * 6 + i];
        Mcn[e] = s;
      }
      __syncthreads();
      for (int e = tid; e < 6 * ND; e += 256) { R[e] = Rn[e]; Mc[e] = Mcn[e]; }
    }
    __syncthreads();
  }
}

__global__ void cvt_kernel(__bf16* __restrict__ dst, const float* __restrict__ src, int n) {
  int i = blockIdx.x * 256 + threadIdx.x;
  if (i < n) dst[i] = (__bf16)src[i];
}

// ---------------- Pass 1: per-segment input aggregate  W = sum Abar^(7-c) Mbar u_c ----------------
__global__ void pass1_kernel(const __bf16* __restrict__ Ubf,
                             const __bf16* __restrict__ AbarBf,  // [128][128]
                             const __bf16* __restrict__ MbarBf,  // [128][192]
                             float* __restrict__ Wseg) {
  __shared__ __align__(16) __bf16 Wbf[B_ * ND];
  __shared__ __align__(16) __bf16 Usm[2][B_ * UC];
  int seg = blockIdx.x;
  int tid = threadIdx.x;
  int wave = tid >> 5, lane = tid & 31;
  int lm = lane & 15, lh = lane >> 4;

  const __bf16* Useg = Ubf + (size_t)seg * CPS * B_ * UC;
  async_copy_chunk(Usm[0], Useg, tid);   // prefetch chunk 0

  for (int c = 0; c < CPS; ++c) {
    __syncthreads();   // all reads of buffer (c+1)&1 (iter c-1) done
    if (c + 1 < CPS) {
      async_copy_chunk(Usm[(c + 1) & 1], Useg + (size_t)(c + 1) * B_ * UC, tid);
      wait_async<6>();  // 6 new outstanding; current buffer's 6 have landed
    } else {
      wait_async<0>();
    }
    __syncthreads();   // LDS visibility across waves
    const __bf16* U = Usm[c & 1];

    v8f acc[4];
#pragma unroll
    for (int mt = 0; mt < 4; ++mt) acc[mt] = (v8f){0, 0, 0, 0, 0, 0, 0, 0};
    // + Mbar * u_c   (K = 192)
    for (int k = 0; k < UC; k += 32) {
      v16bf b = load_b_frag(MbarBf + wave * 16 * UC + k, UC, lane);
#pragma unroll
      for (int mt = 0; mt < 4; ++mt) {
        v16bf a = load_a_frag(U + mt * 16 * UC + k, UC, lane);
        acc[mt] = wmma_bf16(a, b, acc[mt]);
      }
    }
    // + Abar * W     (K = 128)
    if (c > 0) {
      for (int k = 0; k < ND; k += 32) {
        v16bf b = load_b_frag(AbarBf + wave * 16 * ND + k, ND, lane);
#pragma unroll
        for (int mt = 0; mt < 4; ++mt) {
          v16bf a = load_a_frag(Wbf + mt * 16 * ND + k, ND, lane);
          acc[mt] = wmma_bf16(a, b, acc[mt]);
        }
      }
    }
    __syncthreads();
    if (c < CPS - 1) {
#pragma unroll
      for (int mt = 0; mt < 4; ++mt)
#pragma unroll
        for (int r = 0; r < 8; ++r)
          Wbf[(mt * 16 + lh * 8 + r) * ND + wave * 16 + lm] = (__bf16)acc[mt][r];
    } else {
      float* Wout = Wseg + (size_t)seg * B_ * ND;
#pragma unroll
      for (int mt = 0; mt < 4; ++mt)
#pragma unroll
        for (int r = 0; r < 8; ++r)
          Wout[(mt * 16 + lh * 8 + r) * ND + wave * 16 + lm] = acc[mt][r];
    }
  }
}

// ---------------- Pass 2: sequential combine over 64 segment boundaries ----------------
__global__ void pass2_kernel(const float* __restrict__ init,
                             const float* __restrict__ Wseg,
                             const __bf16* __restrict__ AsegBf,  // A^256 [128][128]
                             __bf16* __restrict__ Xsegbf) {
  __shared__ __align__(16) __bf16 Xbf[B_ * ND];
  int wave = threadIdx.x >> 5, lane = threadIdx.x & 31;
  int lm = lane & 15, lh = lane >> 4;

  for (int e = threadIdx.x; e < B_ * ND; e += 256) {
    __bf16 bv = (__bf16)init[e & (ND - 1)];   // broadcast initial_state over batch
    Xbf[e] = bv;
    Xsegbf[e] = bv;
  }
  __syncthreads();

  for (int s = 0; s < NSEG - 1; ++s) {
    v8f acc[4];
#pragma unroll
    for (int mt = 0; mt < 4; ++mt) acc[mt] = (v8f){0, 0, 0, 0, 0, 0, 0, 0};
    for (int k = 0; k < ND; k += 32) {
      v16bf b = load_b_frag(AsegBf + wave * 16 * ND + k, ND, lane);
#pragma unroll
      for (int mt = 0; mt < 4; ++mt) {
        v16bf a = load_a_frag(Xbf + mt * 16 * ND + k, ND, lane);
        acc[mt] = wmma_bf16(a, b, acc[mt]);
      }
    }
    const float* W = Wseg + (size_t)s * B_ * ND;
#pragma unroll
    for (int mt = 0; mt < 4; ++mt)
#pragma unroll
      for (int r = 0; r < 8; ++r)
        acc[mt][r] += W[(mt * 16 + lh * 8 + r) * ND + wave * 16 + lm];
    __syncthreads();
    __bf16* Xout = Xsegbf + (size_t)(s + 1) * B_ * ND;
#pragma unroll
    for (int mt = 0; mt < 4; ++mt)
#pragma unroll
      for (int r = 0; r < 8; ++r) {
        int idx = (mt * 16 + lh * 8 + r) * ND + wave * 16 + lm;
        __bf16 bv = (__bf16)acc[mt][r];
        Xbf[idx] = bv;
        Xout[idx] = bv;
      }
    __syncthreads();
  }
}

// ---------------- Pass 3: per-segment outputs  Y = Cbar X + Gbar u ; X <- Abar X + Mbar u ----------------
__global__ void pass3_kernel(const __bf16* __restrict__ Ubf,
                             const __bf16* __restrict__ Xsegbf,
                             const __bf16* __restrict__ AbarBf,
                             const __bf16* __restrict__ MbarBf,
                             const __bf16* __restrict__ CbarBf,  // [192][128]
                             const __bf16* __restrict__ GbarBf,  // [192][192]
                             const float* __restrict__ omean,
                             const float* __restrict__ ostd,
                             float* __restrict__ out) {
  __shared__ __align__(16) __bf16 Xbf[B_ * ND];
  __shared__ __align__(16) __bf16 Usm[2][B_ * UC];
  int seg = blockIdx.x;
  int tid = threadIdx.x;
  int wave = tid >> 5, lane = tid & 31;
  int lm = lane & 15, lh = lane >> 4;

  const __bf16* Useg = Ubf + (size_t)seg * CPS * B_ * UC;
  async_copy_chunk(Usm[0], Useg, tid);   // prefetch chunk 0 (overlaps Xbf load)

  for (int e = tid; e < B_ * ND; e += 256)
    Xbf[e] = Xsegbf[(size_t)seg * B_ * ND + e];

  for (int c = 0; c < CPS; ++c) {
    __syncthreads();   // prior reads of buffer being overwritten are done
    if (c + 1 < CPS) {
      async_copy_chunk(Usm[(c + 1) & 1], Useg + (size_t)(c + 1) * B_ * UC, tid);
      wait_async<6>();
    } else {
      wait_async<0>();
    }
    __syncthreads();   // LDS visibility (also covers Xbf init on c==0)
    const __bf16* U = Usm[c & 1];
    int chunk = seg * CPS + c;

    // ---- Y = X*Cbar^T + u*Gbar^T : 4x12 tiles, 6 per wave ----
    for (int j = 0; j < 6; ++j) {
      int t = wave + 8 * j;
      int mt = t / 12, nt = t % 12;
      v8f acc = (v8f){0, 0, 0, 0, 0, 0, 0, 0};
      for (int k = 0; k < ND; k += 32) {
        v16bf b = load_b_frag(CbarBf + nt * 16 * ND + k, ND, lane);
        v16bf a = load_a_frag(Xbf + mt * 16 * ND + k, ND, lane);
        acc = wmma_bf16(a, b, acc);
      }
      for (int k = 0; k < UC; k += 32) {
        v16bf b = load_b_frag(GbarBf + nt * 16 * UC + k, UC, lane);
        v16bf a = load_a_frag(U + mt * 16 * UC + k, UC, lane);
        acc = wmma_bf16(a, b, acc);
      }
      int n = nt * 16 + lm;        // 0..191 = 6*t' + o
      int tp = n / 6, o = n - 6 * tp;
      float sc = ostd[o], mn = omean[o];
      int tg = chunk * CHL + tp;
#pragma unroll
      for (int r = 0; r < 8; ++r) {
        int b_ = mt * 16 + lh * 8 + r;
        out[((size_t)b_ * SEQ_ + tg) * 6 + o] = acc[r] * sc + mn;
      }
    }

    // ---- X <- X*Abar^T + u*Mbar^T : 4x8 tiles, wave owns column tile ----
    if (c < CPS - 1) {
      v8f xacc[4];
#pragma unroll
      for (int mt = 0; mt < 4; ++mt) xacc[mt] = (v8f){0, 0, 0, 0, 0, 0, 0, 0};
      for (int k = 0; k < UC; k += 32) {
        v16bf b = load_b_frag(MbarBf + wave * 16 * UC + k, UC, lane);
#pragma unroll
        for (int mt = 0; mt < 4; ++mt) {
          v16bf a = load_a_frag(U + mt * 16 * UC + k, UC, lane);
          xacc[mt] = wmma_bf16(a, b, xacc[mt]);
        }
      }
      for (int k = 0; k < ND; k += 32) {
        v16bf b = load_b_frag(AbarBf + wave * 16 * ND + k, ND, lane);
#pragma unroll
        for (int mt = 0; mt < 4; ++mt) {
          v16bf a = load_a_frag(Xbf + mt * 16 * ND + k, ND, lane);
          xacc[mt] = wmma_bf16(a, b, xacc[mt]);
        }
      }
      __syncthreads();
#pragma unroll
      for (int mt = 0; mt < 4; ++mt)
#pragma unroll
        for (int r = 0; r < 8; ++r)
          Xbf[(mt * 16 + lh * 8 + r) * ND + wave * 16 + lm] = (__bf16)xacc[mt][r];
    }
  }
}

// ---------------- host side ----------------
extern "C" void kernel_launch(void* const* d_in, const int* in_sizes, int n_in,
                              void* d_out, int out_size, void* d_ws, size_t ws_size,
                              hipStream_t stream) {
  (void)in_sizes; (void)n_in; (void)out_size; (void)ws_size;
  const float* in    = (const float*)d_in[0];
  const float* A     = (const float*)d_in[1];
  const float* Bm    = (const float*)d_in[2];
  const float* Cm    = (const float*)d_in[3];
  const float* Dm    = (const float*)d_in[4];
  const float* init  = (const float*)d_in[5];
  const float* imean = (const float*)d_in[6];
  const float* istd  = (const float*)d_in[7];
  const float* omean = (const float*)d_in[8];
  const float* ostd  = (const float*)d_in[9];
  float* out = (float*)d_out;

  char* w = (char*)d_ws;
  auto carve = [&](size_t bytes) -> char* {
    char* p = w;
    w += (bytes + 255) & ~(size_t)255;
    return p;
  };
  __bf16* Ubf    = (__bf16*)carve((size_t)NCHUNK * B_ * UC * 2);
  float*  P2     = (float*)carve(ND * ND * 4);
  float*  P4     = (float*)carve(ND * ND * 4);
  float*  P8     = (float*)carve(ND * ND * 4);
  float*  P16    = (float*)carve(ND * ND * 4);
  float*  P32    = (float*)carve(ND * ND * 4);
  float*  P64    = (float*)carve(ND * ND * 4);
  float*  P128   = (float*)carve(ND * ND * 4);
  float*  P256   = (float*)carve(ND * ND * 4);
  float*  CbarF  = (float*)carve(UC * ND * 4);
  float*  MbarF  = (float*)carve(ND * UC * 4);
  float*  GbarF  = (float*)carve(UC * UC * 4);
  __bf16* AbarBf = (__bf16*)carve(ND * ND * 2);
  __bf16* AsegBf = (__bf16*)carve(ND * ND * 2);
  __bf16* MbarBf = (__bf16*)carve(ND * UC * 2);
  __bf16* CbarBf = (__bf16*)carve(UC * ND * 2);
  __bf16* GbarBf = (__bf16*)carve(UC * UC * 2);
  float*  Wseg   = (float*)carve((size_t)NSEG * B_ * ND * 4);
  __bf16* Xsegbf = (__bf16*)carve((size_t)NSEG * B_ * ND * 2);

  prep_u_kernel<<<2048, 256, 0, stream>>>(in, imean, istd, Ubf);

  mm128_kernel<<<64, 256, 0, stream>>>(P2, A, A);
  mm128_kernel<<<64, 256, 0, stream>>>(P4, P2, P2);
  mm128_kernel<<<64, 256, 0, stream>>>(P8, P4, P4);
  mm128_kernel<<<64, 256, 0, stream>>>(P16, P8, P8);
  mm128_kernel<<<64, 256, 0, stream>>>(P32, P16, P16);
  mm128_kernel<<<64, 256, 0, stream>>>(P64, P32, P32);
  mm128_kernel<<<64, 256, 0, stream>>>(P128, P64, P64);
  mm128_kernel<<<64, 256, 0, stream>>>(P256, P128, P128);

  build_ops_kernel<<<1, 256, 0, stream>>>(A, Bm, Cm, Dm, CbarF, MbarF, GbarF);

  cvt_kernel<<<64, 256, 0, stream>>>(AbarBf, P32, ND * ND);
  cvt_kernel<<<64, 256, 0, stream>>>(AsegBf, P256, ND * ND);
  cvt_kernel<<<96, 256, 0, stream>>>(MbarBf, MbarF, ND * UC);
  cvt_kernel<<<96, 256, 0, stream>>>(CbarBf, CbarF, UC * ND);
  cvt_kernel<<<144, 256, 0, stream>>>(GbarBf, GbarF, UC * UC);

  pass1_kernel<<<NSEG, 256, 0, stream>>>(Ubf, AbarBf, MbarBf, Wseg);
  pass2_kernel<<<1, 256, 0, stream>>>(init, Wseg, AsegBf, Xsegbf);
  pass3_kernel<<<NSEG, 256, 0, stream>>>(Ubf, Xsegbf, AbarBf, MbarBf,
                                         CbarBf, GbarBf, omean, ostd, out);
}
